// SelectiveAttention_77163382440689
// MI455X (gfx1250) — matrix operand order repaired and checked by reading, not compiled, not run on hardware
//
#include <hip/hip_runtime.h>
#include <hip/hip_bf16.h>

typedef __attribute__((ext_vector_type(16))) _Float16 v16h;
typedef __attribute__((ext_vector_type(8)))  _Float16 v8h;
typedef __attribute__((ext_vector_type(4)))  _Float16 v4h;
typedef __attribute__((ext_vector_type(2)))  _Float16 v2h;
typedef __attribute__((ext_vector_type(8)))  float    v8f;

#define T_SEQ   2048
#define N_HEADS 16
#define EMBED   2048
#define HEADD   128
#define LDX     (3 * EMBED + 1)   // 6145 floats per x row
#define BK      64                // keys per chunk

// LDS strides (in halves); all row strides are multiples of 8 halves (16B)
#define KT_STRIDE 72    // Kt[d][key],  128 x 72 halves (144B rows)
#define VT_STRIDE 136   // Vt[kidx][d],  64 x 136 halves (272B rows)
#define P_STRIDE  72    // P[wave][m][kidx], rows of 64 + pad (144B rows)

// ds_swizzle_b32 XOR mode: offset = {xor[14:10], or[9:5], and[4:0]} ; EXEC all-1 here
#define SWIZ_XOR(v, m) \
    __int_as_float(__builtin_amdgcn_ds_swizzle(__float_as_int(v), ((m) << 10) | 0x1f))

__device__ __forceinline__ v8f wmma_f16(v16h a, v16h b, v8f c) {
    return __builtin_amdgcn_wmma_f32_16x16x32_f16(
        false, a, false, b, (short)0, c, false, false);
}

// One 64-key chunk. K-index permutation: kidx = 4*ncol + ks <-> key = 16*ks + ncol
// (applied to P columns and V rows alike; WMMA contraction is permutation-invariant).
template <bool MASK>
__device__ __forceinline__ void process_chunk(
    const v16h* __restrict__ qa, const v16h& ones,
    v8f* __restrict__ Oacc, v8f& Lacc, float* __restrict__ mrow,
    const _Float16* __restrict__ Kt, const _Float16* __restrict__ Vt,
    _Float16* __restrict__ Pw, int lane, int kk0, int qw0)
{
    const int mrow_lo = (lane >> 4) * 8;
    const int ncol    = lane & 15;
    const int k0      = (lane >> 4) * 8;

    // ---- S = Q * K^T : four 16x16 tiles across 64 keys (double-buffered B) ----
    auto loadB = [&](v16h* dst, int c) {
        const v8h* kr = (const v8h*)&Kt[(32 * c + lane) * KT_STRIDE];
        #pragma unroll
        for (int ks = 0; ks < 4; ++ks) {
            #pragma unroll
            for (int j = 0; j < 8; ++j) {
                dst[ks][j] = Kt[(32 * c + k0 + j) * KT_STRIDE + 16 * ks + ncol]; dst[ks][8 + j] = Kt[(32 * c + 16 + k0 + j) * KT_STRIDE + 16 * ks + ncol];
            }
        }
    };

    v8f s[4];
    #pragma unroll
    for (int ks = 0; ks < 4; ++ks) s[ks] = (v8f)0.0f;

    v16h kbuf[2][4];
    loadB(kbuf[0], 0);
    #pragma unroll
    for (int c = 0; c < 4; ++c) {
        if (c < 3) loadB(kbuf[(c + 1) & 1], c + 1);
        #pragma unroll
        for (int ks = 0; ks < 4; ++ks)
            s[ks] = wmma_f16(qa[c], kbuf[c & 1][ks], s[ks]);
    }

    // ---- online softmax in log2 units (row sums accumulated by WMMA via Lacc) ----
    float corr_arr[8];
    #pragma unroll
    for (int r = 0; r < 8; ++r) {
        const int qg = qw0 + mrow_lo + r;
        float a[4];
        #pragma unroll
        for (int ks = 0; ks < 4; ++ks) {
            a[ks] = s[ks][r];
            if (MASK) a[ks] = ((kk0 + 16 * ks + ncol) <= qg) ? a[ks] : -1.0e30f;
        }
        float mx = fmaxf(fmaxf(a[0], a[1]), fmaxf(a[2], a[3]));
        mx = fmaxf(mx, SWIZ_XOR(mx, 1));
        mx = fmaxf(mx, SWIZ_XOR(mx, 2));
        mx = fmaxf(mx, SWIZ_XOR(mx, 4));
        mx = fmaxf(mx, SWIZ_XOR(mx, 8));

        const float mnew = fmaxf(mrow[r], mx);
        corr_arr[r] = __builtin_exp2f(mrow[r] - mnew);
        mrow[r] = mnew;

        v4h pv;
        #pragma unroll
        for (int ks = 0; ks < 4; ++ks)
            pv[ks] = (_Float16)__builtin_exp2f(a[ks] - mnew);
        // packed store: kidx = 4*ncol + ks  (one b64 per row)
        *(v4h*)(Pw + (mrow_lo + r) * P_STRIDE + 4 * ncol) = pv;
    }

    // ---- rescale accumulators with packed f32 muls ----
    v8f cv;
    #pragma unroll
    for (int r = 0; r < 8; ++r) cv[r] = corr_arr[r];
    #pragma unroll
    for (int nc = 0; nc < 8; ++nc) Oacc[nc] *= cv;
    Lacc *= cv;

    // ---- re-read P as two 16x32 A-fragments (same-wave DS is in-order) ----
    v16h pa0, pa1;
    {
        const _Float16* pr = Pw + (lane & 15) * P_STRIDE;
        #pragma unroll
        for (int j = 0; j < 8; ++j) {
            pa0[j]     = pr[k0 + j];       pa0[8 + j] = pr[16 + k0 + j];
            pa1[j]     = pr[32 + k0 + j];  pa1[8 + j] = pr[48 + k0 + j];
        }
    }

    // every column of P*ones is the row sum -> Lacc stays lane-replicated
    Lacc = wmma_f16(pa0, ones, Lacc);
    Lacc = wmma_f16(pa1, ones, Lacc);

    // ---- O += P * V over 8 D-chunks of 16 (double-buffered B) ----
    auto loadV = [&](v16h* dst, int nc) {
        const v8h* v0 = (const v8h*)&Vt[lane * VT_STRIDE + nc * 16];
        const v8h* v1 = (const v8h*)&Vt[(32 + lane) * VT_STRIDE + nc * 16];
        #pragma unroll
        for (int j = 0; j < 8; ++j) {
            dst[0][j] = Vt[(k0 + j) * VT_STRIDE + nc * 16 + ncol]; dst[0][8 + j] = Vt[(16 + k0 + j) * VT_STRIDE + nc * 16 + ncol];
            dst[1][j] = Vt[(32 + k0 + j) * VT_STRIDE + nc * 16 + ncol]; dst[1][8 + j] = Vt[(48 + k0 + j) * VT_STRIDE + nc * 16 + ncol];
        }
    };

    v16h vbuf[2][2];
    loadV(vbuf[0], 0);
    #pragma unroll
    for (int nc = 0; nc < 8; ++nc) {
        if (nc < 7) loadV(vbuf[(nc + 1) & 1], nc + 1);
        Oacc[nc] = wmma_f16(pa0, vbuf[nc & 1][0], Oacc[nc]);
        Oacc[nc] = wmma_f16(pa1, vbuf[nc & 1][1], Oacc[nc]);
    }
}

__global__ __launch_bounds__(256)
void SelectiveAttention_77163382440689_kernel(const float* __restrict__ x,
                                              float* __restrict__ y) {
    __shared__ __align__(16) _Float16 Kt[128 * KT_STRIDE];       // 18432 B
    __shared__ __align__(16) _Float16 Vt[BK * VT_STRIDE];        // 17408 B
    __shared__ __align__(16) _Float16 Pl[8 * 16 * P_STRIDE];     // 18432 B

    const int qblk = blockIdx.x;
    const int h    = blockIdx.y;
    const int tid  = threadIdx.x;
    const int wave = tid >> 5;
    const int lane = tid & 31;

    const int q0  = qblk * 128;
    const int qw0 = q0 + wave * 16;

    const int mrow_lo = (lane >> 4) * 8;
    const int ncol    = lane & 15;
    const int k0      = (lane >> 4) * 8;

    // ---- load Q tile scaled by (1/sqrt(D)) * log2(e) -> softmax in log2 units ----
    const float qscale = 0.08838834764831845f * 1.4426950408889634f;
    const float* qrow = x + (size_t)(qw0 + (lane & 15)) * LDX + h * HEADD;
    v16h qa[4];
    #pragma unroll
    for (int c = 0; c < 4; ++c) {
        #pragma unroll
        for (int j = 0; j < 8; ++j) {
            qa[c][j]     = (_Float16)(qrow[32 * c + k0 + j] * qscale);
            qa[c][8 + j] = (_Float16)(qrow[32 * c + 16 + k0 + j] * qscale);
        }
    }

    v16h ones;
    #pragma unroll
    for (int j = 0; j < 16; ++j) ones[j] = (_Float16)1.0f;

    v8f   Oacc[8], Lacc = (v8f)0.0f;
    float mrow[8];
    #pragma unroll
    for (int nc = 0; nc < 8; ++nc) Oacc[nc] = (v8f)0.0f;
    #pragma unroll
    for (int r = 0; r < 8; ++r) mrow[r] = -1.0e30f;

    // staging: each thread owns two adjacent keys (kpair, kpair+1) x 16 dims
    //  -> Kt scatter becomes aligned b32 stores on consecutive banks
    const int kpair = (tid & 31) * 2;       // even key 0..62
    const int dhalf = (tid >> 5) * 16;      // 0..112
    const int vidx0 = 4 * (kpair & 15) + (kpair >> 4);        // permuted K-index
    const int vidx1 = 4 * ((kpair + 1) & 15) + ((kpair + 1) >> 4);
    _Float16* Pw = &Pl[wave * 16 * P_STRIDE];

    const int nkc = 2 * qblk + 2;

    for (int kc = 0; kc < nkc; ++kc) {
        const int kk0 = kc * BK;

        __syncthreads();

        {
            const float* k0p = x + (size_t)(kk0 + kpair) * LDX + EMBED     + h * HEADD + dhalf;
            const float* v0p = x + (size_t)(kk0 + kpair) * LDX + 2 * EMBED + h * HEADD + dhalf;
            #pragma unroll
            for (int j = 0; j < 16; ++j) {
                v2h pr;
                pr[0] = (_Float16)k0p[j];
                pr[1] = (_Float16)k0p[LDX + j];
                *(v2h*)&Kt[(dhalf + j) * KT_STRIDE + kpair] = pr;
            }
            #pragma unroll
            for (int j = 0; j < 16; ++j) {
                Vt[vidx0 * VT_STRIDE + dhalf + j] = (_Float16)v0p[j];
                Vt[vidx1 * VT_STRIDE + dhalf + j] = (_Float16)v0p[LDX + j];
            }
        }
        __syncthreads();

        if (kc + 1 < nkc) {
            const float* kn = x + (size_t)(kk0 + BK + kpair) * LDX + EMBED + h * HEADD + dhalf;
            __builtin_prefetch(kn, 0, 3);
            __builtin_prefetch(kn + LDX, 0, 3);
            __builtin_prefetch(kn + EMBED, 0, 3);
            __builtin_prefetch(kn + LDX + EMBED, 0, 3);
        }

        if (kk0 <= qw0 + 15) {                       // wave-uniform branches
            if (kk0 + (BK - 1) <= qw0)
                process_chunk<false>(qa, ones, Oacc, Lacc, mrow, Kt, Vt, Pw, lane, kk0, qw0);
            else
                process_chunk<true >(qa, ones, Oacc, Lacc, mrow, Kt, Vt, Pw, lane, kk0, qw0);
        }
    }

    // ---- normalize and store (Lacc columns all hold the row sum) ----
    v8f rv;
    #pragma unroll
    for (int r = 0; r < 8; ++r) rv[r] = 1.0f / Lacc[r];

    #pragma unroll
    for (int nc = 0; nc < 8; ++nc) {
        v8f o = Oacc[nc] * rv;
        #pragma unroll
        for (int r = 0; r < 8; ++r) {
            const int t = qw0 + mrow_lo + r;
            const int d = nc * 16 + ncol;
            y[(size_t)t * EMBED + h * HEADD + d] = o[r];
        }
    }
}

extern "C" void kernel_launch(void* const* d_in, const int* in_sizes, int n_in,
                              void* d_out, int out_size, void* d_ws, size_t ws_size,
                              hipStream_t stream) {
    const float* x = (const float*)d_in[0];
    float* y = (float*)d_out;
    (void)in_sizes; (void)n_in; (void)out_size; (void)d_ws; (void)ws_size;
    dim3 grid(T_SEQ / 128, N_HEADS);
    dim3 block(256);
    SelectiveAttention_77163382440689_kernel<<<grid, block, 0, stream>>>(x, y);
}
